// Lfm2MoeSparseMoeBlock_43963285242543
// MI455X (gfx1250) — matrix operand order repaired and checked by reading, not compiled, not run on hardware
//
#include <hip/hip_runtime.h>
#include <hip/hip_bf16.h>
#include <math.h>

// Problem constants (match reference)
#define DD   2048   // hidden
#define FF   1024   // intermediate
#define EE   16     // experts
#define TT   2048   // tokens (B*S)

typedef __attribute__((ext_vector_type(16))) __bf16 v16bf;
typedef __attribute__((ext_vector_type(8)))  __bf16 v8bf;
typedef __attribute__((ext_vector_type(4)))  __bf16 v4bf;
typedef __attribute__((ext_vector_type(8)))  float  v8f;

// LDS strides (elements), padded for conflict-free ds_load_b128 fragment reads
#define XS_STRIDE 2056   // 16 rows of X  (bf16)
#define HS_STRIDE 1048   // 16 rows of H  (bf16)
#define WS_STRIDE 40     // transposed weight slabs (bf16)

#define HS_BYTES   (16 * HS_STRIDE * 2)           // 33536
#define XS_BYTES   (16 * XS_STRIDE * 2)           // 65792
#define WGU_BYTES  (128 * WS_STRIDE * 2)          // 10240 (per matrix)
#define STG1_BYTES (XS_BYTES + 2 * WGU_BYTES)     // 86272
#define WD_BYTES   (2048 * WS_STRIDE * 2)         // 163840
#define STG_MAX    (WD_BYTES > STG1_BYTES ? WD_BYTES : STG1_BYTES)
#define SMEM_MAIN  (HS_BYTES + STG_MAX)           // 197376 B < 320 KB WGP limit

__device__ __forceinline__ __bf16 tob(float f) { return (__bf16)f; }

__device__ __forceinline__ float fast_rcp(float x) {
#if __has_builtin(__builtin_amdgcn_rcpf)
  return __builtin_amdgcn_rcpf(x);     // raw v_rcp_f32: plenty for bf16 output
#else
  return 1.0f / x;
#endif
}

__device__ __forceinline__ v16bf load_frag16(const __bf16* p) {
  v8bf lo = *(const v8bf*)(p);
  v8bf hi = *(const v8bf*)(p + 16);
  return __builtin_shufflevector(lo, hi, 0,1,2,3,4,5,6,7,8,9,10,11,12,13,14,15);
}

// transpose 4x4 fp32 -> 4 columns of 4 packed bf16 at dst, dst+WS, dst+2WS, dst+3WS
__device__ __forceinline__ void store_t4(__bf16* dst, float4 a, float4 b,
                                         float4 c, float4 d) {
  *(v4bf*)(dst)                 = (v4bf){tob(a.x), tob(b.x), tob(c.x), tob(d.x)};
  *(v4bf*)(dst + WS_STRIDE)     = (v4bf){tob(a.y), tob(b.y), tob(c.y), tob(d.y)};
  *(v4bf*)(dst + 2 * WS_STRIDE) = (v4bf){tob(a.z), tob(b.z), tob(c.z), tob(d.z)};
  *(v4bf*)(dst + 3 * WS_STRIDE) = (v4bf){tob(a.w), tob(b.w), tob(c.w), tob(d.w)};
}

// ---------------------------------------------------------------------------
// Kernel 1: routing
// ---------------------------------------------------------------------------
__global__ void moe_route(const float* __restrict__ x,
                          const float* __restrict__ w_gate,
                          const float* __restrict__ expert_bias,
                          int*   __restrict__ counts,
                          int*   __restrict__ tok_ids,
                          float* __restrict__ tok_w) {
  __shared__ float red[256 * EE];
  __shared__ float logits[EE];

  const int t = blockIdx.x;
  const float* xr = x + (size_t)t * DD;

  float acc[EE];
#pragma unroll
  for (int e = 0; e < EE; ++e) acc[e] = 0.f;

  for (int d = threadIdx.x; d < DD; d += 256) {
    const float xv = xr[d];
    const float* wr = w_gate + (size_t)d * EE;
#pragma unroll
    for (int e = 0; e < EE; ++e) acc[e] += xv * wr[e];
  }
#pragma unroll
  for (int e = 0; e < EE; ++e) red[threadIdx.x * EE + e] = acc[e];
  __syncthreads();

  if (threadIdx.x < EE) {
    float s = 0.f;
    for (int i = 0; i < 256; ++i) s += red[i * EE + threadIdx.x];
    logits[threadIdx.x] = s;
  }
  __syncthreads();

  if (threadIdx.x == 0) {
    float m = logits[0];
#pragma unroll
    for (int e = 1; e < EE; ++e) m = fmaxf(m, logits[e]);
    float p[EE];
    float den = 0.f;
#pragma unroll
    for (int e = 0; e < EE; ++e) { p[e] = __expf(logits[e] - m); den += p[e]; }
    const float inv = 1.f / den;   // routing stays full fp32 precision
#pragma unroll
    for (int e = 0; e < EE; ++e) p[e] = p[e] * inv + expert_bias[e];
    int i0 = 0;
#pragma unroll
    for (int e = 1; e < EE; ++e) if (p[e] > p[i0]) i0 = e;
    int i1 = (i0 == 0) ? 1 : 0;
#pragma unroll
    for (int e = 0; e < EE; ++e) if (e != i0 && p[e] > p[i1]) i1 = e;
    float s0 = p[i0], s1 = p[i1];
    const float ninv = 1.f / (s0 + s1 + 1e-20f);
    s0 *= ninv; s1 *= ninv;
    int pos0 = atomicAdd(&counts[i0], 1);
    tok_ids[i0 * TT + pos0] = t;  tok_w[i0 * TT + pos0] = s0;
    int pos1 = atomicAdd(&counts[i1], 1);
    tok_ids[i1 * TT + pos1] = t;  tok_w[i1 * TT + pos1] = s1;
  }
}

// ---------------------------------------------------------------------------
// Kernel 2: fused expert MLP with software-pipelined slab loads
// ---------------------------------------------------------------------------
__global__ void __launch_bounds__(256, 1)
moe_experts(const float* __restrict__ x,
            const float* __restrict__ w_g,
            const float* __restrict__ w_u,
            const float* __restrict__ w_d,
            const int*   __restrict__ counts,
            const int*   __restrict__ tok_ids,
            const float* __restrict__ tok_w,
            float*       __restrict__ y) {
  const int e    = blockIdx.y;
  const int tile = blockIdx.x;
  const int n    = counts[e];
  const int row0 = tile * 16;
  if (row0 >= n) return;                 // uniform per block, before any barrier
  const int rows = (n - row0 < 16) ? (n - row0) : 16;

  __shared__ __align__(16) char smem[SMEM_MAIN];
  __shared__ int   tokS[16];
  __shared__ float twS[16];

  __bf16* Hs  = (__bf16*)smem;                          // [16][HS_STRIDE]
  __bf16* Xs  = (__bf16*)(smem + HS_BYTES);             // [16][XS_STRIDE]
  __bf16* WgT = Xs + 16 * XS_STRIDE;                    // [128][WS_STRIDE]
  __bf16* WuT = WgT + 128 * WS_STRIDE;                  // [128][WS_STRIDE]
  __bf16* WdT = (__bf16*)(smem + HS_BYTES);             // [2048][WS_STRIDE] (aliases stage-1 area)

  const int tid  = threadIdx.x;
  const int wv   = tid >> 5;
  const int lane = tid & 31;
  const int half = lane >> 4;
  const int l15  = lane & 15;

  // ---- token metadata + X tile (fp32 -> bf16) into LDS -------------------
  if (tid < 16) {
    tokS[tid] = (tid < rows) ? tok_ids[e * TT + row0 + tid] : 0;
    twS[tid]  = (tid < rows) ? tok_w [e * TT + row0 + tid] : 0.f;
  }
  {
    const int r = tid >> 4;
    const int c0 = (tid & 15) * 4;
    const int trow = (r < rows) ? tok_ids[e * TT + row0 + r] : 0;
    const float* xr = x + (size_t)trow * DD;
    for (int c = c0; c < DD; c += 64) {
      float4 v = make_float4(0.f, 0.f, 0.f, 0.f);
      if (r < rows) v = *(const float4*)(xr + c);
      *(v4bf*)(Xs + r * XS_STRIDE + c) = (v4bf){tob(v.x), tob(v.y), tob(v.z), tob(v.w)};
    }
  }
  __syncthreads();

  const int kg4 = (tid >> 5) * 4;          // this thread's 4 k-rows (0..28)
  const int nb1 = (tid & 31) * 4;          // stage-1: 4 n-cols
  const int nb2 = (tid & 31) * 64;         // stage-2: 64 n-cols

  // ---- Stage 1: H = silu(X@Wg) * (X@Wu) ----------------------------------
  for (int fc = 0; fc < FF; fc += 128) {
    v8f accg = {};
    v8f accu = {};
    const float* gcol = w_g + ((size_t)e * DD + kg4) * FF + fc + nb1;
    const float* ucol = w_u + ((size_t)e * DD + kg4) * FF + fc + nb1;

    // prologue prefetch (k0 = 0)
    float4 pg0 = *(const float4*)(gcol);
    float4 pg1 = *(const float4*)(gcol + FF);
    float4 pg2 = *(const float4*)(gcol + 2 * FF);
    float4 pg3 = *(const float4*)(gcol + 3 * FF);
    float4 pu0 = *(const float4*)(ucol);
    float4 pu1 = *(const float4*)(ucol + FF);
    float4 pu2 = *(const float4*)(ucol + 2 * FF);
    float4 pu3 = *(const float4*)(ucol + 3 * FF);

    for (int k0 = 0; k0 < DD; k0 += 32) {
      __syncthreads();   // previous slab fully consumed
      store_t4(WgT + nb1 * WS_STRIDE + kg4, pg0, pg1, pg2, pg3);
      store_t4(WuT + nb1 * WS_STRIDE + kg4, pu0, pu1, pu2, pu3);
      if (k0 + 32 < DD) {
        const float* gn = gcol + (size_t)(k0 + 32) * FF;
        const float* un = ucol + (size_t)(k0 + 32) * FF;
        pg0 = *(const float4*)(gn);
        pg1 = *(const float4*)(gn + FF);
        pg2 = *(const float4*)(gn + 2 * FF);
        pg3 = *(const float4*)(gn + 3 * FF);
        pu0 = *(const float4*)(un);
        pu1 = *(const float4*)(un + FF);
        pu2 = *(const float4*)(un + 2 * FF);
        pu3 = *(const float4*)(un + 3 * FF);
      }
      __syncthreads();   // slab visible
      v16bf A  = load_frag16(Xs + l15 * XS_STRIDE + k0 + half * 8);
      v16bf Bg = load_frag16(WgT + (wv * 16 + l15) * WS_STRIDE + half * 8);
      v16bf Bu = load_frag16(WuT + (wv * 16 + l15) * WS_STRIDE + half * 8);
      accg = __builtin_amdgcn_wmma_f32_16x16x32_bf16(false, A, false, Bg, (short)0, accg, false, false);
      accu = __builtin_amdgcn_wmma_f32_16x16x32_bf16(false, A, false, Bu, (short)0, accu, false, false);
    }
    // h = g * sigmoid(g) * u via raw v_rcp_f32 (bf16 output precision)
#pragma unroll
    for (int j = 0; j < 8; ++j) {
      const float g = accg[j];
      const float u = accu[j];
      const float h = g * fast_rcp(1.f + __expf(-g)) * u;
      const int rr = half * 8 + j;
      Hs[rr * HS_STRIDE + fc + wv * 16 + l15] = tob(h);
    }
  }
  __syncthreads();   // Hs complete; stage-1 slab region reusable

  // ---- Stage 2: Y = H @ Wd; wave wv owns columns [wv*256, wv*256+256) ----
  v8f acc[16];
#pragma unroll
  for (int i = 0; i < 16; ++i) acc[i] = (v8f){};

  const float* drow = w_d + ((size_t)e * FF + kg4) * DD + nb2;

  // pipeline at half-tile granularity: 32 of this thread's 64 columns
  // (4 rows x 8 float4 = 32 VGPR-float4) prefetched one K-step ahead.
  float4 buf[32];
#pragma unroll
  for (int r4 = 0; r4 < 4; ++r4)
#pragma unroll
    for (int q = 0; q < 8; ++q)
      buf[r4 * 8 + q] = *(const float4*)(drow + r4 * DD + q * 4);

  for (int k0 = 0; k0 < FF; k0 += 32) {
    const float* cur = drow + (size_t)k0 * DD;
    __syncthreads();   // previous slab fully consumed
    // first half (cols nb2 .. nb2+31) from the prefetch buffer
#pragma unroll
    for (int q = 0; q < 8; ++q)
      store_t4(WdT + (nb2 + q * 4) * WS_STRIDE + kg4,
               buf[0 * 8 + q], buf[1 * 8 + q], buf[2 * 8 + q], buf[3 * 8 + q]);
    // second half (cols nb2+32 .. nb2+63) loaded and stored directly
#pragma unroll
    for (int q = 0; q < 8; ++q) {
      float4 a = *(const float4*)(cur + 0 * DD + 32 + q * 4);
      float4 b = *(const float4*)(cur + 1 * DD + 32 + q * 4);
      float4 c = *(const float4*)(cur + 2 * DD + 32 + q * 4);
      float4 d = *(const float4*)(cur + 3 * DD + 32 + q * 4);
      store_t4(WdT + (nb2 + 32 + q * 4) * WS_STRIDE + kg4, a, b, c, d);
    }
    // prefetch next K-step's first half; completes under the WMMAs
    if (k0 + 32 < FF) {
      const float* nxt = drow + (size_t)(k0 + 32) * DD;
#pragma unroll
      for (int r4 = 0; r4 < 4; ++r4)
#pragma unroll
        for (int q = 0; q < 8; ++q)
          buf[r4 * 8 + q] = *(const float4*)(nxt + r4 * DD + q * 4);
    }
    __syncthreads();   // slab visible
    v16bf A = load_frag16(Hs + l15 * HS_STRIDE + k0 + half * 8);
#pragma unroll
    for (int tl = 0; tl < 16; ++tl) {
      const int ncol = wv * 256 + tl * 16 + l15;
      v16bf B = load_frag16(WdT + ncol * WS_STRIDE + half * 8);
      acc[tl] = __builtin_amdgcn_wmma_f32_16x16x32_bf16(false, A, false, B, (short)0, acc[tl], false, false);
    }
  }

  // ---- weighted scatter-add into y ---------------------------------------
#pragma unroll
  for (int j = 0; j < 8; ++j) {
    const int rr = half * 8 + j;
    if (rr < rows) {
      const int   t  = tokS[rr];
      const float wt = twS[rr];
#pragma unroll
      for (int tl = 0; tl < 16; ++tl) {
        const int col = wv * 256 + tl * 16 + l15;
        atomicAdd(&y[(size_t)t * DD + col], wt * acc[tl][j]);
      }
    }
  }
}

// ---------------------------------------------------------------------------
// Host launcher
// ---------------------------------------------------------------------------
extern "C" void kernel_launch(void* const* d_in, const int* in_sizes, int n_in,
                              void* d_out, int out_size, void* d_ws, size_t ws_size,
                              hipStream_t stream) {
  const float* x           = (const float*)d_in[0];   // [T, D]
  const float* w_gate      = (const float*)d_in[1];   // [D, E]
  const float* w_g         = (const float*)d_in[2];   // [E, D, F]
  const float* w_u         = (const float*)d_in[3];   // [E, D, F]
  const float* w_d         = (const float*)d_in[4];   // [E, F, D]
  const float* expert_bias = (const float*)d_in[5];   // [E]
  float* y = (float*)d_out;                           // [T, D]

  char* ws = (char*)d_ws;
  int*   counts  = (int*)ws;                           // E ints
  int*   tok_ids = (int*)(ws + 256);                   // E*T ints
  float* tok_w   = (float*)(ws + 256 + EE * TT * 4);   // E*T floats

  hipMemsetAsync(counts, 0, EE * sizeof(int), stream);
  hipMemsetAsync(y, 0, (size_t)out_size * sizeof(float), stream);

  moe_route<<<TT, 256, 0, stream>>>(x, w_gate, expert_bias, counts, tok_ids, tok_w);

  dim3 grid(TT / 16, EE);
  moe_experts<<<grid, 256, 0, stream>>>(x, w_g, w_u, w_d, counts, tok_ids, tok_w, y);

  (void)in_sizes; (void)n_in; (void)ws_size;
}
